// SAMBLOCK_51092930953605
// MI455X (gfx1250) — compile-verified, matrix operand order checked by reading, not run on hardware
//
#include <hip/hip_runtime.h>
#include <math.h>

// ---------------------------------------------------------------------------
// Types / tiling config
// ---------------------------------------------------------------------------
typedef __attribute__((ext_vector_type(16))) _Float16 v16h;
typedef __attribute__((ext_vector_type(8)))  _Float16 v8h;
typedef __attribute__((ext_vector_type(8)))  float    v8f;

#define BM   128
#define BN   128
#define BKK  32
#define LPAD 8
#define LDSW (BKK + LPAD)

constexpr int       kB   = 8;
constexpr int       kL   = 9216;           // 96*96
constexpr int       kC   = 512;
constexpr long long kBL  = 73728;          // B*L
constexpr int       kHID = 2048;
constexpr long long kLC  = (long long)kL * kC;

// Epilogue selectors
constexpr int EPI_F16          = 0;  // outH = acc (+ bias[n] if bias)
constexpr int EPI_F16_GELU     = 1;  // outH = gelu(acc + bias[n])
constexpr int EPI_REPROJ       = 2;  // outF = acc + bias[m] + xn^T + x   (writes x1)
constexpr int EPI_F32_BIAS_RES = 3;  // outF = acc + bias[n] + res1[idx]  (fc2 + residual)

struct GemmArgs {
  const _Float16* A;   // A[m][k]  element at aOuter*bo + aInner*bi + m*aRow + k*aK
  const _Float16* Bw;  // B[n][k]  (i.e. out = A @ B^T)
  long long aOuter, aInner, aRow, aK;
  long long bOuter, bInner, bRow, bK;
  int M, N, K, zDiv;
  const float*    bias;
  const float*    res1;  // f32 residual (flat, same index as out)
  const _Float16* resH;  // f16 residual (reproj: xn, read transposed)
  float*    outF;
  _Float16* outH;
  long long oOuter, oInner, oRow, oCol;
};

// ---------------------------------------------------------------------------
// Staging: global -> regs -> LDS  (128x32 f16 tile, 256 threads)
// ---------------------------------------------------------------------------
union Stage {
  v8h      v[2];
  _Float16 h[16];
};

__device__ __forceinline__ void load_tile(Stage& st, const _Float16* __restrict__ base,
                                          long long rowStride, long long kStride,
                                          int bound, int rowOff, int k0, int kEnd,
                                          int t, bool contig) {
  if (contig) {
    // 128 rows x 32 halves = 512 segments of 16B; 2 per thread
#pragma unroll
    for (int i = 0; i < 2; ++i) {
      int seg = t + i * 256;
      int r   = seg >> 2;
      int sc  = (seg & 3) * 8;
      int gr  = rowOff + r;
      v8h val = {};
      if (gr < bound) {
        const _Float16* p = base + (long long)gr * rowStride + (k0 + sc);
        val = *(const v8h*)p;
        // Pull the tile-after-next into L2/WGP$ (global_prefetch_b8, no LOADcnt).
        if (k0 + sc + BKK < kEnd)
          __builtin_prefetch((const void*)(p + BKK), 0, 3);
      }
      st.v[i] = val;
    }
  } else {
    // element-wise gather (K-strided operand, e.g. k_sm^T / val^T)
#pragma unroll
    for (int i = 0; i < 16; ++i) {
      int e  = t + i * 256;
      int r  = e >> 5;
      int kk = e & 31;
      int gr = rowOff + r;
      _Float16 val = (_Float16)0.0f;
      if (gr < bound)
        val = base[(long long)gr * rowStride + (long long)(k0 + kk) * kStride];
      st.h[i] = val;
    }
  }
}

__device__ __forceinline__ void store_tile(_Float16 (*tile)[LDSW], const Stage& st,
                                           int t, bool contig) {
  if (contig) {
#pragma unroll
    for (int i = 0; i < 2; ++i) {
      int seg = t + i * 256;
      int r   = seg >> 2;
      int sc  = (seg & 3) * 8;
      *(v8h*)&tile[r][sc] = st.v[i];
    }
  } else {
#pragma unroll
    for (int i = 0; i < 16; ++i) {
      int e = t + i * 256;
      tile[e >> 5][e & 31] = st.h[i];
    }
  }
}

// A/B fragment for v_wmma_f32_16x16x32_f16, per ISA 16-bit 16x32 layout:
// lane (0..15): M=lane, holds K = 8*kgrp + {0..7} (lo) and +16 (hi); kgrp = lane>>4.
__device__ __forceinline__ v16h load_frag(const _Float16* row, int kgrp) {
  v8h lo = *(const v8h*)(row + 8 * kgrp);
  v8h hi = *(const v8h*)(row + 8 * kgrp + 16);
  return __builtin_shufflevector(lo, hi, 0, 1, 2, 3, 4, 5, 6, 7,
                                         8, 9, 10, 11, 12, 13, 14, 15);
}

__device__ __forceinline__ void compute_step(const _Float16 (*As)[LDSW],
                                             const _Float16 (*Bs)[LDSW],
                                             int waveM, int waveN, int lane,
                                             v8f acc[2][4]) {
  const int kgrp = lane >> 4;
  const int rl   = lane & 15;
  v16h fa[2], fb[4];
#pragma unroll
  for (int mi = 0; mi < 2; ++mi)
    fa[mi] = load_frag(&As[waveM * 32 + mi * 16 + rl][0], kgrp);
#pragma unroll
  for (int ni = 0; ni < 4; ++ni)
    fb[ni] = load_frag(&Bs[waveN * 64 + ni * 16 + rl][0], kgrp);
#pragma unroll
  for (int mi = 0; mi < 2; ++mi)
#pragma unroll
    for (int ni = 0; ni < 4; ++ni)
      acc[mi][ni] = __builtin_amdgcn_wmma_f32_16x16x32_f16(
          false, fa[mi], false, fb[ni], (short)0, acc[mi][ni], false, false);
}

// ---------------------------------------------------------------------------
// Generic WMMA GEMM: out = A @ B^T (+epilogue).  Block = 256 thr = 8 waves.
// Wave grid 4(M) x 2(N); each wave: 2x4 tiles of 16x16 -> 32x64 region.
// ---------------------------------------------------------------------------
template <int EPI>
__global__ __launch_bounds__(256) void wmma_gemm(GemmArgs ga) {
  __shared__ _Float16 As[2][BM][LDSW];
  __shared__ _Float16 Bs[2][BN][LDSW];

  const int t     = threadIdx.x;
  const int wave  = t >> 5;
  const int lane  = t & 31;
  const int waveM = wave & 3;
  const int waveN = wave >> 2;

  const int bo = (int)blockIdx.z / ga.zDiv;
  const int bi = (int)blockIdx.z % ga.zDiv;

  const _Float16* Abase = ga.A  + (long long)bo * ga.aOuter + (long long)bi * ga.aInner;
  const _Float16* Bbase = ga.Bw + (long long)bo * ga.bOuter + (long long)bi * ga.bInner;

  const int blockM = blockIdx.y * BM;
  const int blockN = blockIdx.x * BN;
  const bool contigA = (ga.aK == 1);
  const bool contigB = (ga.bK == 1);

  v8f acc[2][4] = {};

  const int KT = ga.K / BKK;

  Stage ra, rb;
  load_tile(ra, Abase, ga.aRow, ga.aK, ga.M, blockM, 0, ga.K, t, contigA);
  load_tile(rb, Bbase, ga.bRow, ga.bK, ga.N, blockN, 0, ga.K, t, contigB);
  store_tile(As[0], ra, t, contigA);
  store_tile(Bs[0], rb, t, contigB);
  __syncthreads();

  for (int kt = 0; kt < KT; ++kt) {
    const int  cur = kt & 1;
    const bool has = (kt + 1 < KT);
    if (has) {
      load_tile(ra, Abase, ga.aRow, ga.aK, ga.M, blockM, (kt + 1) * BKK, ga.K, t, contigA);
      load_tile(rb, Bbase, ga.bRow, ga.bK, ga.N, blockN, (kt + 1) * BKK, ga.K, t, contigB);
    }
    compute_step(As[cur], Bs[cur], waveM, waveN, lane, acc);
    if (has) {
      store_tile(As[cur ^ 1], ra, t, contigA);
      store_tile(Bs[cur ^ 1], rb, t, contigB);
    }
    __syncthreads();
  }

  // Epilogue. C/D layout: VGPR r -> M = r + 8*(lane>=16); N = lane & 15.
  const int nL = lane & 15;
  const int hs = lane >> 4;
#pragma unroll
  for (int mi = 0; mi < 2; ++mi) {
#pragma unroll
    for (int ni = 0; ni < 4; ++ni) {
      const int nG = blockN + waveN * 64 + ni * 16 + nL;
      if (nG >= ga.N) continue;
      const int mB = blockM + waveM * 32 + mi * 16 + hs * 8;
#pragma unroll
      for (int r = 0; r < 8; ++r) {
        const int mG = mB + r;
        if (mG >= ga.M) continue;
        const float a = acc[mi][ni][r];
        const long long oIdx = (long long)bo * ga.oOuter + (long long)bi * ga.oInner +
                               (long long)mG * ga.oRow + (long long)nG * ga.oCol;
        if (EPI == EPI_F16) {
          float vv = a + (ga.bias ? ga.bias[nG] : 0.0f);
          ga.outH[oIdx] = (_Float16)vv;
        } else if (EPI == EPI_F16_GELU) {
          float vv = a + ga.bias[nG];
          vv = 0.5f * vv * (1.0f + erff(vv * 0.70710678118654752f));
          ga.outH[oIdx] = (_Float16)vv;
        } else if (EPI == EPI_REPROJ) {
          // out[bo][m=o][n=l] (flat B,C,L) += br[o] + xn[bo][l][o] + x[flat]
          float r0 = (float)ga.resH[(long long)bo * ga.oOuter + (long long)nG * kC + mG];
          ga.outF[oIdx] = a + ga.bias[mG] + r0 + ga.res1[oIdx];
        } else {  // EPI_F32_BIAS_RES
          ga.outF[oIdx] = a + ga.bias[nG] + ga.res1[oIdx];
        }
      }
    }
  }
}

// ---------------------------------------------------------------------------
// LayerNorm over C=512 per row; writes f16. Block 256 (2 elems/thread).
// ---------------------------------------------------------------------------
__global__ __launch_bounds__(256) void ln_rows(const float* __restrict__ in,
                                               const float* __restrict__ w,
                                               const float* __restrict__ b,
                                               _Float16* __restrict__ outH) {
  const long long row = blockIdx.x;
  const int t = threadIdx.x;
  const float* p = in + row * kC;
  float x0 = p[t], x1 = p[t + 256];

  __shared__ float red[256];
  red[t] = x0 + x1;
  __syncthreads();
  for (int s = 128; s > 0; s >>= 1) { if (t < s) red[t] += red[t + s]; __syncthreads(); }
  const float mean = red[0] * (1.0f / 512.0f);
  __syncthreads();

  const float d0 = x0 - mean, d1 = x1 - mean;
  red[t] = d0 * d0 + d1 * d1;
  __syncthreads();
  for (int s = 128; s > 0; s >>= 1) { if (t < s) red[t] += red[t + s]; __syncthreads(); }
  const float rstd = rsqrtf(red[0] * (1.0f / 512.0f) + 1e-5f);

  outH[row * kC + t]       = (_Float16)(d0 * rstd * w[t] + b[t]);
  outH[row * kC + t + 256] = (_Float16)(d1 * rstd * w[t + 256] + b[t + 256]);
}

// ---------------------------------------------------------------------------
// q softmax: per (b,l,head) over 64 contiguous channels, in place (f16).
// One wave32 per head-chunk; 8 waves per block cover one row of 512.
// ---------------------------------------------------------------------------
__global__ __launch_bounds__(256) void softmax_q(_Float16* __restrict__ q) {
  const long long row = blockIdx.x;
  const int w = threadIdx.x >> 5, lane = threadIdx.x & 31;
  _Float16* p = q + row * kC + w * 64;
  float a = (float)p[lane];
  float b = (float)p[lane + 32];
  float mx = fmaxf(a, b);
#pragma unroll
  for (int off = 16; off > 0; off >>= 1) mx = fmaxf(mx, __shfl_xor(mx, off, 32));
  float ea = __expf(a - mx), eb = __expf(b - mx);
  float s = ea + eb;
#pragma unroll
  for (int off = 16; off > 0; off >>= 1) s += __shfl_xor(s, off, 32);
  const float inv = 1.0f / s;
  p[lane]      = (_Float16)(ea * inv);
  p[lane + 32] = (_Float16)(eb * inv);
}

// ---------------------------------------------------------------------------
// k softmax: over L=9216 per (b, channel), in place (f16). Block per column.
// ---------------------------------------------------------------------------
__global__ __launch_bounds__(256) void softmax_k(_Float16* __restrict__ k) {
  const long long col = blockIdx.x;          // b*512 + c
  const long long b   = col >> 9;
  const long long c   = col & 511;
  _Float16* base = k + b * kLC + c;
  const int t = threadIdx.x;

  float vbuf[36];
  float mx = -3.4e38f;
#pragma unroll
  for (int i = 0; i < 36; ++i) {
    vbuf[i] = (float)base[(long long)(t + i * 256) * kC];
    mx = fmaxf(mx, vbuf[i]);
  }
  __shared__ float red[256];
  red[t] = mx;
  __syncthreads();
  for (int s = 128; s > 0; s >>= 1) { if (t < s) red[t] = fmaxf(red[t], red[t + s]); __syncthreads(); }
  mx = red[0];
  __syncthreads();

  float sum = 0.0f;
#pragma unroll
  for (int i = 0; i < 36; ++i) { vbuf[i] = __expf(vbuf[i] - mx); sum += vbuf[i]; }
  red[t] = sum;
  __syncthreads();
  for (int s = 128; s > 0; s >>= 1) { if (t < s) red[t] += red[t + s]; __syncthreads(); }
  const float inv = 1.0f / red[0];
#pragma unroll
  for (int i = 0; i < 36; ++i)
    base[(long long)(t + i * 256) * kC] = (_Float16)(vbuf[i] * inv);
}

// ---------------------------------------------------------------------------
// f32 -> f16 weight convert
// ---------------------------------------------------------------------------
__global__ void cvt_f32_f16(const float* __restrict__ s, _Float16* __restrict__ d, int n) {
  int i = blockIdx.x * 256 + threadIdx.x;
  if (i < n) d[i] = (_Float16)s[i];
}

// ---------------------------------------------------------------------------
// Host orchestration
// ---------------------------------------------------------------------------
extern "C" void kernel_launch(void* const* d_in, const int* in_sizes, int n_in,
                              void* d_out, int out_size, void* d_ws, size_t ws_size,
                              hipStream_t stream) {
  (void)in_sizes; (void)n_in; (void)out_size; (void)ws_size;
  const float* x    = (const float*)d_in[0];
  const float* v    = (const float*)d_in[1];
  const float* ln1w = (const float*)d_in[4];
  const float* ln1b = (const float*)d_in[5];
  const float* lnvw = (const float*)d_in[6];
  const float* lnvb = (const float*)d_in[7];
  const float* ln2w = (const float*)d_in[8];
  const float* ln2b = (const float*)d_in[9];
  const float* wq   = (const float*)d_in[10];
  const float* bq   = (const float*)d_in[11];
  const float* wk   = (const float*)d_in[12];
  const float* bk   = (const float*)d_in[13];
  const float* wv   = (const float*)d_in[14];
  const float* bv   = (const float*)d_in[15];
  const float* wr   = (const float*)d_in[16];
  const float* br   = (const float*)d_in[17];
  const float* f1w  = (const float*)d_in[18];
  const float* f1b  = (const float*)d_in[19];
  const float* f2w  = (const float*)d_in[20];
  const float* f2b  = (const float*)d_in[21];
  float* out = (float*)d_out;

  char* wsb = (char*)d_ws;
  size_t off = 0;
  auto alloc = [&](size_t bytes) -> void* {
    void* p = wsb + off;
    off += (bytes + 255) & ~(size_t)255;
    return p;
  };

  float*    x1    = (float*)alloc((size_t)kBL * kC * sizeof(float));
  _Float16* xn_h  = (_Float16*)alloc((size_t)kBL * kC * 2);
  _Float16* vn_h  = (_Float16*)alloc((size_t)kBL * kC * 2);  // later re-used: attn_t, fc1-out start
  _Float16* q_h   = (_Float16*)alloc((size_t)kBL * kC * 2);
  _Float16* k_h   = (_Float16*)alloc((size_t)kBL * kC * 2);
  _Float16* val_h = (_Float16*)alloc((size_t)kBL * kC * 2);
  _Float16* ctxT  = (_Float16*)alloc((size_t)kB * 8 * 64 * 64 * 2);
  _Float16* wq_h  = (_Float16*)alloc((size_t)512 * 512 * 2);
  _Float16* wk_h  = (_Float16*)alloc((size_t)512 * 512 * 2);
  _Float16* wv_h  = (_Float16*)alloc((size_t)512 * 512 * 2);
  _Float16* wr_h  = (_Float16*)alloc((size_t)512 * 512 * 2);
  _Float16* f1w_h = (_Float16*)alloc((size_t)2048 * 512 * 2);
  _Float16* f2w_h = (_Float16*)alloc((size_t)512 * 2048 * 2);

  _Float16* attn_t = vn_h;  // (B, L, Cv) f16 — vn dead after val GEMM
  _Float16* h1     = vn_h;  // (B, L, 2048) f16 aliases vn,q,k,val (4x75.5MB = 302MB)
  _Float16* ln2h   = xn_h;  // xn dead after reproj

  // --- weights -> f16 ---
  cvt_f32_f16<<<(512 * 512 + 255) / 256, 256, 0, stream>>>(wq, wq_h, 512 * 512);
  cvt_f32_f16<<<(512 * 512 + 255) / 256, 256, 0, stream>>>(wk, wk_h, 512 * 512);
  cvt_f32_f16<<<(512 * 512 + 255) / 256, 256, 0, stream>>>(wv, wv_h, 512 * 512);
  cvt_f32_f16<<<(512 * 512 + 255) / 256, 256, 0, stream>>>(wr, wr_h, 512 * 512);
  cvt_f32_f16<<<(2048 * 512 + 255) / 256, 256, 0, stream>>>(f1w, f1w_h, 2048 * 512);
  cvt_f32_f16<<<(512 * 2048 + 255) / 256, 256, 0, stream>>>(f2w, f2w_h, 512 * 2048);

  // --- LayerNorms on x and v ---
  ln_rows<<<(unsigned)kBL, 256, 0, stream>>>(x, ln1w, ln1b, xn_h);
  ln_rows<<<(unsigned)kBL, 256, 0, stream>>>(v, lnvw, lnvb, vn_h);

  const dim3 blk(256, 1, 1);

  // --- q / k / val projections: (BL x 512) @ (512 x 512)^T ---
  {
    GemmArgs ga{};
    ga.A = xn_h; ga.aRow = 512; ga.aK = 1;
    ga.Bw = wq_h; ga.bRow = 512; ga.bK = 1;
    ga.M = (int)kBL; ga.N = 512; ga.K = 512; ga.zDiv = 1;
    ga.bias = bq; ga.outH = q_h; ga.oRow = 512; ga.oCol = 1;
    wmma_gemm<EPI_F16><<<dim3(4, 576, 1), blk, 0, stream>>>(ga);
    ga.A = vn_h; ga.Bw = wk_h; ga.bias = bk; ga.outH = k_h;
    wmma_gemm<EPI_F16><<<dim3(4, 576, 1), blk, 0, stream>>>(ga);
    ga.Bw = wv_h; ga.bias = bv; ga.outH = val_h;
    wmma_gemm<EPI_F16><<<dim3(4, 576, 1), blk, 0, stream>>>(ga);
  }

  // --- softmaxes (in place) ---
  softmax_q<<<(unsigned)kBL, 256, 0, stream>>>(q_h);
  softmax_k<<<(unsigned)(kB * 512), 256, 0, stream>>>(k_h);

  // --- context^T per (b,h): out[k][v]-> stored as [v][k]; K = L = 9216 ---
  {
    GemmArgs ga{};
    ga.A = k_h;  ga.aOuter = kLC; ga.aInner = 64; ga.aRow = 1; ga.aK = 512;
    ga.Bw = val_h; ga.bOuter = kLC; ga.bInner = 64; ga.bRow = 1; ga.bK = 512;
    ga.M = 64; ga.N = 64; ga.K = kL; ga.zDiv = 8;
    ga.outH = ctxT; ga.oOuter = 8 * 4096; ga.oInner = 4096; ga.oRow = 1; ga.oCol = 64;
    wmma_gemm<EPI_F16><<<dim3(1, 1, 64), blk, 0, stream>>>(ga);
  }

  // --- attended: attn_t[b][l][h*64+v] = q_sm[b][l][h][:] . ctxT[b][h][v][:] ---
  {
    GemmArgs ga{};
    ga.A = q_h;  ga.aOuter = kLC; ga.aInner = 64; ga.aRow = 512; ga.aK = 1;
    ga.Bw = ctxT; ga.bOuter = 8 * 4096; ga.bInner = 4096; ga.bRow = 64; ga.bK = 1;
    ga.M = kL; ga.N = 64; ga.K = 64; ga.zDiv = 8;
    ga.outH = attn_t; ga.oOuter = kLC; ga.oInner = 64; ga.oRow = 512; ga.oCol = 1;
    wmma_gemm<EPI_F16><<<dim3(1, 72, 64), blk, 0, stream>>>(ga);
  }

  // --- reproj + channel-major residuals -> x1 (flat f32) ---
  {
    GemmArgs ga{};
    ga.A = wr_h; ga.aRow = 512; ga.aK = 1;                       // shared weights
    ga.Bw = attn_t; ga.bOuter = kLC; ga.bRow = 512; ga.bK = 1;   // per-batch agg^T
    ga.M = 512; ga.N = kL; ga.K = 512; ga.zDiv = 1;              // z = batch
    ga.bias = br; ga.resH = xn_h; ga.res1 = x;
    ga.outF = x1; ga.oOuter = kLC; ga.oRow = kL; ga.oCol = 1;
    wmma_gemm<EPI_REPROJ><<<dim3(72, 4, 8), blk, 0, stream>>>(ga);
  }

  // --- LN2 ---
  ln_rows<<<(unsigned)kBL, 256, 0, stream>>>(x1, ln2w, ln2b, ln2h);

  // --- fc1 + GELU ---
  {
    GemmArgs ga{};
    ga.A = ln2h; ga.aRow = 512; ga.aK = 1;
    ga.Bw = f1w_h; ga.bRow = 512; ga.bK = 1;
    ga.M = (int)kBL; ga.N = kHID; ga.K = 512; ga.zDiv = 1;
    ga.bias = f1b; ga.outH = h1; ga.oRow = kHID; ga.oCol = 1;
    wmma_gemm<EPI_F16_GELU><<<dim3(16, 576, 1), blk, 0, stream>>>(ga);
  }

  // --- fc2 + bias + residual -> out ---
  {
    GemmArgs ga{};
    ga.A = h1; ga.aRow = kHID; ga.aK = 1;
    ga.Bw = f2w_h; ga.bRow = kHID; ga.bK = 1;
    ga.M = (int)kBL; ga.N = 512; ga.K = kHID; ga.zDiv = 1;
    ga.bias = f2b; ga.res1 = x1;
    ga.outF = out; ga.oRow = 512; ga.oCol = 1;
    wmma_gemm<EPI_F32_BIAS_RES><<<dim3(4, 576, 1), blk, 0, stream>>>(ga);
  }
}
